// TEAM_pos_9637906612271
// MI455X (gfx1250) — matrix operand order repaired and checked by reading, not compiled, not run on hardware
//
#include <hip/hip_runtime.h>
#include <hip/hip_bf16.h>
#include <math.h>

typedef __attribute__((ext_vector_type(16))) _Float16 v16h;
typedef __attribute__((ext_vector_type(8)))  _Float16 v8h;
typedef __attribute__((ext_vector_type(8)))  float    v8f;

#define T_FRAMES 16
#define CDIM     256
#define KSTEPS   8            // 256 / 32
#define HALVES_PER_KSTEP 512  // 32 lanes * 16 halves

// ---------------------------------------------------------------------------
// Kernel 1: L2-normalize each length-256 vector and emit f16 pre-swizzled into
// the exact per-lane WMMA register layout (A-layout for q, B-layout for s).
// One wave per vector. Lane i owns 8 contiguous channels.
// ---------------------------------------------------------------------------
__global__ __launch_bounds__(256) void normalize_pack_kernel(
    const float* __restrict__ src, _Float16* __restrict__ dst,
    int nvec, int bmode)
{
    const int lane = threadIdx.x & 31;
    int v = blockIdx.x * 8 + (threadIdx.x >> 5);
    if (v >= nvec) v = nvec - 1;          // clamp (uniform per wave), dup write ok
    const int t   = v & (T_FRAMES - 1);   // frame index -> WMMA row/col
    const int obj = v >> 4;               // q or s index

    const int kstep = lane >> 2;          // 0..7
    const int part  = lane & 3;           // which 8-channel chunk of this kstep

    const float* p = src + (size_t)v * CDIM + kstep * 32 + part * 8;
    float4 x0 = *(const float4*)(p);
    float4 x1 = *(const float4*)(p + 4);

    float ss = x0.x*x0.x + x0.y*x0.y + x0.z*x0.z + x0.w*x0.w
             + x1.x*x1.x + x1.y*x1.y + x1.z*x1.z + x1.w*x1.w;
    #pragma unroll
    for (int off = 16; off >= 1; off >>= 1) ss += __shfl_xor(ss, off);

    const float scale = 1.0f / fmaxf(sqrtf(ss), 1e-12f);

    v8h h;
    h[0] = (_Float16)(x0.x * scale); h[1] = (_Float16)(x0.y * scale);
    h[2] = (_Float16)(x0.z * scale); h[3] = (_Float16)(x0.w * scale);
    h[4] = (_Float16)(x1.x * scale); h[5] = (_Float16)(x1.y * scale);
    h[6] = (_Float16)(x1.z * scale); h[7] = (_Float16)(x1.w * scale);

    // A-matrix 16x32 f16 layout (ISA 7.12.2): lanes 0-15 hold K 0..7,16..23;
    // lanes 16-31 hold K 8..15,24..31 (row = lane&15).
    // B-matrix 32x16 layout: same K order as A: lanes 0-15 hold K 0..7,16..23;
    // lanes 16-31 hold K 8..15,24..31 (col = lane&15).
    int wl, off8;
    if (bmode == 0) { wl = t + ((part & 1) << 4); off8 = (part >> 1) * 8; }
    else            { wl = t + ((part & 1) << 4); off8 = (part >> 1) * 8; }

    size_t addr = (((size_t)obj * KSTEPS + kstep) * 32 + wl) * 16 + off8;
    *(v8h*)(dst + addr) = h;
}

// ---------------------------------------------------------------------------
// Kernel 2: fused tile-GEMM (WMMA f16 -> f32) + bidirectional OTAM soft-DTW.
// One wave per (q,s) pair. 8 waves / block, one LDS 16x16 tile per wave.
// ---------------------------------------------------------------------------
#define TSTR 19  // LDS row stride: 18*r mod 64 injective for r<16 -> no conflicts

__global__ __launch_bounds__(256) void team_gemm_dp_kernel(
    const _Float16* __restrict__ qA, const _Float16* __restrict__ sB,
    float* __restrict__ out, int npairs, int Ns)
{
    __shared__ float tiles[8][16 * TSTR];

    const int lane = threadIdx.x & 31;
    const int w    = threadIdx.x >> 5;
    int pair = blockIdx.x * 8 + w;
    if (pair >= npairs) pair = npairs - 1;   // uniform per wave; dup write ok
    const int q = pair / Ns;
    const int s = pair - q * Ns;

    // ---- GEMM: sim[16][16] = q_tile(16x256) . s_tile(256x16) via 8 WMMAs ----
    const v16h* aPtr = (const v16h*)qA + (size_t)q * (KSTEPS * 32) + lane;
    const v16h* bPtr = (const v16h*)sB + (size_t)s * (KSTEPS * 32) + lane;
    v8f acc = {};
    #pragma unroll
    for (int ks = 0; ks < KSTEPS; ++ks) {
        v16h a = aPtr[ks * 32];
        v16h b = bPtr[ks * 32];
        acc = __builtin_amdgcn_wmma_f32_16x16x32_f16(
            false, a, false, b, (short)0, acc, false, false);
    }

    // ---- scatter dists = 1 - sim into LDS per C/D layout ----
    float* tile = tiles[w];
    {
        const int n  = lane & 15;
        const int mb = (lane >> 4) * 8;
        #pragma unroll
        for (int r = 0; r < 8; ++r)
            tile[(mb + r) * TSTR + n] = 1.0f - acc[r];
    }
    __syncthreads();

    // ---- OTAM DP, anti-diagonal wavefront ----
    // lanes 0-15: direction 0 (rows of dists); lanes 16-31: direction 1
    // (rows of dists^T). Padded cost d[l][m], m in [0,17]: d[l][0]=d[l][17]=0,
    // d[l][m]=dists[l][m-1] (dir0) or dists[m-1][l] (dir1).
    const float lbda  = 0.3f;
    const float inv_l = 1.0f / 0.3f;
    const int   row   = lane & 15;
    const int   dir   = lane >> 4;

    float cur = 0.0f, prev = 0.0f;
    for (int t = 0; t <= 32; ++t) {
        // lane l-1 runs one m-step ahead: its cur = cum[l-1][m], prev = cum[l-1][m-1]
        float up_cur  = __shfl_up(cur, 1);
        float up_prev = __shfl_up(prev, 1);
        const int m = t - row;
        if (m >= 0 && m <= 17) {
            float dval = 0.0f;
            if (m >= 1 && m <= 16)
                dval = dir ? tile[(m - 1) * TSTR + row]
                           : tile[row * TSTR + (m - 1)];
            float val;
            if (m == 0) {
                val = 0.0f;                       // cum[l][0] = 0 (row0: d[0][0]=0)
            } else if (row == 0) {
                val = cur + dval;                 // row 0: running cumsum
            } else {
                const float a0 = up_prev;         // cum[l-1][m-1]
                const float b0 = cur;             // cum[l][m-1]
                float m0 = fminf(a0, b0);
                float ssum;
                if (m == 1 || m == 17) {          // include cum[l-1][m]
                    const float c0 = up_cur;
                    m0   = fminf(m0, c0);
                    ssum = __expf(-(a0 - m0) * inv_l)
                         + __expf(-(b0 - m0) * inv_l)
                         + __expf(-(c0 - m0) * inv_l);
                } else {
                    ssum = __expf(-(a0 - m0) * inv_l)
                         + __expf(-(b0 - m0) * inv_l);
                }
                val = dval + m0 - lbda * __logf(ssum);
            }
            prev = cur;
            cur  = val;
        }
    }

    const float r0 = __shfl(cur, 15);  // dir0 final: cum[15][17]
    const float r1 = __shfl(cur, 31);  // dir1 final
    if (lane == 0) out[(size_t)q * Ns + s] = r0 + r1;
}

// ---------------------------------------------------------------------------
extern "C" void kernel_launch(void* const* d_in, const int* in_sizes, int n_in,
                              void* d_out, int out_size, void* d_ws, size_t ws_size,
                              hipStream_t stream)
{
    const float* traj_q = (const float*)d_in[0];
    const float* traj_s = (const float*)d_in[1];
    float*       out    = (float*)d_out;

    const int Nq = in_sizes[0] / (T_FRAMES * CDIM);   // 1024
    const int Ns = in_sizes[1] / (T_FRAMES * CDIM);   // 64

    _Float16* qA = (_Float16*)d_ws;                               // Nq*8*512 halves
    _Float16* sB = qA + (size_t)Nq * KSTEPS * HALVES_PER_KSTEP;   // Ns*8*512 halves

    const int nvq = Nq * T_FRAMES;
    const int nvs = Ns * T_FRAMES;
    normalize_pack_kernel<<<(nvq + 7) / 8, 256, 0, stream>>>(traj_q, qA, nvq, 0);
    normalize_pack_kernel<<<(nvs + 7) / 8, 256, 0, stream>>>(traj_s, sB, nvs, 1);

    const int npairs = Nq * Ns;
    team_gemm_dp_kernel<<<(npairs + 7) / 8, 256, 0, stream>>>(qA, sB, out, npairs, Ns);
}
